// GPT_52201032515650
// MI455X (gfx1250) — matrix operand order repaired and checked by reading, not compiled, not run on hardware
//
#include <hip/hip_runtime.h>
#include <cstdint>

typedef unsigned short ushort_t;
typedef __bf16  v16bf __attribute__((ext_vector_type(16)));
typedef float   v8f   __attribute__((ext_vector_type(8)));
typedef unsigned int v8u __attribute__((ext_vector_type(8)));
typedef unsigned int v4u __attribute__((ext_vector_type(4)));

#define KLB __launch_bounds__(256)

// ---- helpers -------------------------------------------------------------

static __device__ __forceinline__ ushort_t f2bf(float f) {
  union { float f; unsigned int u; } v; v.f = f;
  unsigned int u = v.u;
  unsigned int r = u + 0x7FFFu + ((u >> 16) & 1u);   // round-to-nearest-even
  return (ushort_t)(r >> 16);
}

static __device__ __forceinline__ v16bf mkfrag(v4u a, v4u b) {
  v8u t;
  t[0]=a[0]; t[1]=a[1]; t[2]=a[2]; t[3]=a[3];
  t[4]=b[0]; t[5]=b[1]; t[6]=b[2]; t[7]=b[3];
  return __builtin_bit_cast(v16bf, t);
}

static __device__ __forceinline__ v8f wmma_bf16(v16bf a, v16bf b, v8f c) {
  return __builtin_amdgcn_wmma_f32_16x16x32_bf16(false, a, false, b, (short)0, c,
                                                 false, false);
}

// CDNA5 async memory->LDS copy (ASYNCcnt-tracked, no VGPR roundtrip).
// lds_off = low 32 bits of the generic pointer into __shared__ (LDS aperture
// maps addr[31:0] -> LDS byte offset per ISA 10.2).
static __device__ __forceinline__ void async_load_b128(unsigned lds_off,
                                                       const void* gaddr) {
  asm volatile("global_load_async_to_lds_b128 %0, %1, off"
               :: "v"(lds_off), "v"(gaddr) : "memory");
}

static __device__ __forceinline__ void wait_async0() {
  asm volatile("s_wait_asynccnt 0" ::: "memory");
}

// ---- elementwise kernels -------------------------------------------------

__global__ KLB void cast_kernel(const float* __restrict__ in,
                                ushort_t* __restrict__ out, long n) {
  long i = (long)blockIdx.x * 256 + threadIdx.x;
  if (i < n) out[i] = f2bf(in[i]);
}

// in [K][N] fp32 -> out [N][K] bf16 (weight transpose+cast for Bt layout)
__global__ KLB void castT_kernel(const float* __restrict__ in,
                                 ushort_t* __restrict__ out, int K, int N) {
  long i = (long)blockIdx.x * 256 + threadIdx.x;
  long total = (long)K * N;
  if (i >= total) return;
  int n = (int)(i / K), k = (int)(i % K);
  out[i] = f2bf(in[(size_t)k * N + n]);
}

__global__ KLB void embed_kernel(const int* __restrict__ idx,
                                 const float* __restrict__ wte,
                                 const float* __restrict__ wpe,
                                 float* __restrict__ x) {
  size_t i = (size_t)blockIdx.x * 256 + threadIdx.x;   // < 4096*768
  int bt = (int)(i / 768), e = (int)(i % 768);
  int t = bt & 1023;
  int tok = idx[bt];
  x[i] = wte[(size_t)tok * 768 + e] + wpe[(size_t)t * 768 + e];
}

// split qkv fp32 [4096,2304] -> q,k bf16 [B,H,T,64]; v transposed bf16 [B,H,64,T]
__global__ KLB void qkvsplit_kernel(const float* __restrict__ qkv,
                                    ushort_t* __restrict__ qb,
                                    ushort_t* __restrict__ kb,
                                    ushort_t* __restrict__ vtb) {
  size_t i = (size_t)blockIdx.x * 256 + threadIdx.x;   // < 4096*768
  int bt = (int)(i / 768), c = (int)(i % 768);
  int hh = c >> 6, d = c & 63;
  int b = bt >> 10, t = bt & 1023;
  size_t base = (size_t)bt * 2304;
  int bh = b * 12 + hh;
  qb[((size_t)bh * 1024 + t) * 64 + d] = f2bf(qkv[base + c]);
  kb[((size_t)bh * 1024 + t) * 64 + d] = f2bf(qkv[base + 768 + c]);
  vtb[((size_t)bh * 64 + d) * 1024 + t] = f2bf(qkv[base + 1536 + c]);
}

// ---- layernorm: one row of 768 per block, bf16 output --------------------

__global__ KLB void ln_kernel(const float* __restrict__ x,
                              const float* __restrict__ scale,
                              const float* __restrict__ bias,
                              ushort_t* __restrict__ out) {
  __shared__ float red[256];
  int row = blockIdx.x;
  const float* xr = x + (size_t)row * 768;
  float v[3], s = 0.f;
#pragma unroll
  for (int i = 0; i < 3; i++) { v[i] = xr[threadIdx.x + i * 256]; s += v[i]; }
  red[threadIdx.x] = s; __syncthreads();
  for (int off = 128; off > 0; off >>= 1) {
    if ((int)threadIdx.x < off) red[threadIdx.x] += red[threadIdx.x + off];
    __syncthreads();
  }
  float mu = red[0] * (1.0f / 768.0f);
  __syncthreads();
  s = 0.f;
#pragma unroll
  for (int i = 0; i < 3; i++) { float d = v[i] - mu; s += d * d; }
  red[threadIdx.x] = s; __syncthreads();
  for (int off = 128; off > 0; off >>= 1) {
    if ((int)threadIdx.x < off) red[threadIdx.x] += red[threadIdx.x + off];
    __syncthreads();
  }
  float rstd = rsqrtf(red[0] * (1.0f / 768.0f) + 1e-5f);
  ushort_t* orow = out + (size_t)row * 768;
#pragma unroll
  for (int i = 0; i < 3; i++) {
    int c = threadIdx.x + i * 256;
    orow[c] = f2bf((v[i] - mu) * rstd * scale[c] + bias[c]);
  }
}

// ---- WMMA GEMM: C[M,N] = A[M,K](bf16) x Bt[N,K](bf16) + bias, epilogue ---
// block = 256 threads (8 waves), tile 128x128x32. All dims divide evenly.
// Tiles are streamed memory->LDS with GLOBAL_LOAD_ASYNC_TO_LDS_B128 and
// double-buffered so the 8-WMMA block hides the prefetch latency.

__global__ KLB void gemm_kernel(const ushort_t* __restrict__ A,
                                const ushort_t* __restrict__ Bt,
                                const float* __restrict__ bias,
                                const float* __restrict__ resid,
                                float* __restrict__ outF,
                                ushort_t* __restrict__ outB,
                                int Mdim, int Ndim, int Kdim, int gelu) {
  (void)Mdim;
  __shared__ ushort_t As[2][128 * 32];
  __shared__ ushort_t Bs[2][128 * 32];
  const int tid = threadIdx.x;
  const int lane = tid & 31;
  const int wid = tid >> 5;
  const int lo = lane & 15, hi = lane >> 4;
  const int wm = wid & 3, wn = wid >> 2;        // 4 (M) x 2 (N) wave grid
  const int bm = blockIdx.y * 128, bn = blockIdx.x * 128;

  v8f acc[2][4] = {};

  const int lrow = tid >> 1;                    // 0..127
  const int lcol = (tid & 1) * 16;              // 0 or 16
  const ushort_t* Arow = A + (size_t)(bm + lrow) * Kdim + lcol;
  const ushort_t* Brow = Bt + (size_t)(bn + lrow) * Kdim + lcol;
  const unsigned laA[2] = {(unsigned)(size_t)&As[0][lrow * 32 + lcol],
                           (unsigned)(size_t)&As[1][lrow * 32 + lcol]};
  const unsigned laB[2] = {(unsigned)(size_t)&Bs[0][lrow * 32 + lcol],
                           (unsigned)(size_t)&Bs[1][lrow * 32 + lcol]};

  auto issue_tile = [&](int kk, int buf) {
    const ushort_t* ga = Arow + kk * 32;
    const ushort_t* gb = Brow + kk * 32;
    async_load_b128(laA[buf],       ga);
    async_load_b128(laA[buf] + 16u, ga + 8);
    async_load_b128(laB[buf],       gb);
    async_load_b128(laB[buf] + 16u, gb + 8);
  };

  const int ntiles = Kdim >> 5;
  issue_tile(0, 0);
  wait_async0();
  __syncthreads();

  for (int kk = 0; kk < ntiles; kk++) {
    const int cur = kk & 1;
    if (kk + 1 < ntiles) issue_tile(kk + 1, cur ^ 1);

    v16bf af[2], bfr[4];
#pragma unroll
    for (int mt = 0; mt < 2; mt++) {
      int r = wm * 32 + mt * 16 + lo;           // A fragment: row m = lane&15
      af[mt] = mkfrag(*(const v4u*)&As[cur][r * 32 + hi * 8],
                      *(const v4u*)&As[cur][r * 32 + 16 + hi * 8]);
    }
#pragma unroll
    for (int nt = 0; nt < 4; nt++) {
      int r = wn * 64 + nt * 16 + lo;           // B fragment: col n = lane&15
      bfr[nt] = mkfrag(*(const v4u*)&Bs[cur][r * 32 + hi * 8],
                       *(const v4u*)&Bs[cur][r * 32 + 16 + hi * 8]);
    }
#pragma unroll
    for (int mt = 0; mt < 2; mt++)
#pragma unroll
      for (int nt = 0; nt < 4; nt++)
        acc[mt][nt] = wmma_bf16(af[mt], bfr[nt], acc[mt][nt]);

    wait_async0();          // prefetched tile landed in LDS
    __syncthreads();        // publish to all waves; prior reads retired
  }

  // epilogue: C layout row m = r + 8*hi, col n = lane&15
#pragma unroll
  for (int mt = 0; mt < 2; mt++) {
#pragma unroll
    for (int nt = 0; nt < 4; nt++) {
      int gcol = bn + wn * 64 + nt * 16 + lo;
      float bv = bias ? bias[gcol] : 0.0f;
#pragma unroll
      for (int r = 0; r < 8; r++) {
        int grow = bm + wm * 32 + mt * 16 + r + 8 * hi;
        float v = acc[mt][nt][r] + bv;
        if (gelu) {
          float x3 = v * v * v;
          v = 0.5f * v * (1.0f + tanhf(0.7978845608028654f * (v + 0.044715f * x3)));
        }
        size_t o = (size_t)grow * Ndim + gcol;
        if (resid) v += resid[o];
        if (outF) outF[o] = v;
        if (outB) outB[o] = f2bf(v);
      }
    }
  }
}

// ---- flash attention: one wave per (b,h, 16-row q tile), online softmax --

__global__ KLB void attn_kernel(const ushort_t* __restrict__ qb,
                                const ushort_t* __restrict__ kb,
                                const ushort_t* __restrict__ vtb,
                                ushort_t* __restrict__ yb) {
  __shared__ ushort_t Pbuf[8][16 * 64];         // per-wave P staging
  const int lane = threadIdx.x & 31, wid = threadIdx.x >> 5;
  const int lo = lane & 15, hi = lane >> 4;
  int gw = blockIdx.x * 8 + wid;                // 0..3071
  int qt = gw & 63;
  int bh = gw >> 6;                             // b*H + h, 0..47
  int b = bh / 12, hh = bh % 12;
  int q0 = qt * 16;

  const ushort_t* qbase = qb + (size_t)bh * 1024 * 64;
  const ushort_t* kbase = kb + (size_t)bh * 1024 * 64;
  const ushort_t* vbase = vtb + (size_t)bh * 64 * 1024;

  // Q fragments (K-dim = d = 64 -> two 16x32 A fragments), cached in regs
  const v4u* qr = (const v4u*)(qbase + (size_t)(q0 + lo) * 64);
  v16bf aq0 = mkfrag(qr[hi], qr[2 + hi]);       // d 0..31
  v16bf aq1 = mkfrag(qr[4 + hi], qr[6 + hi]);   // d 32..63

  float mrow[8], lrow[8];
  v8f O[4] = {};
#pragma unroll
  for (int r = 0; r < 8; r++) { mrow[r] = -1e30f; lrow[r] = 0.f; }

  int ktmax = qt >> 2;                          // causal: 64-wide k tiles
  for (int kt = 0; kt <= ktmax; kt++) {
    float s[4][8];
#pragma unroll
    for (int nt = 0; nt < 4; nt++) {
      int tk = kt * 64 + nt * 16 + lo;          // B fragment: n = k-row index
      const v4u* kr = (const v4u*)(kbase + (size_t)tk * 64);
      v16bf kf0 = mkfrag(kr[hi], kr[2 + hi]);          // d 0..31
      v16bf kf1 = mkfrag(kr[4 + hi], kr[6 + hi]);  // d 32..63
      v8f acc = {};
      acc = wmma_bf16(aq0, kf0, acc);
      acc = wmma_bf16(aq1, kf1, acc);
      int col = kt * 64 + nt * 16 + lo;
#pragma unroll
      for (int r = 0; r < 8; r++) {
        int row = q0 + r + 8 * hi;
        float sv = acc[r] * 0.125f;             // 1/sqrt(64)
        s[nt][r] = (col <= row) ? sv : -1e30f;
      }
    }
    float alpha[8], mnew[8];
#pragma unroll
    for (int r = 0; r < 8; r++) {
      float mx = fmaxf(fmaxf(s[0][r], s[1][r]), fmaxf(s[2][r], s[3][r]));
      for (int m = 1; m < 16; m <<= 1) mx = fmaxf(mx, __shfl_xor(mx, m, 32));
      mnew[r] = fmaxf(mrow[r], mx);
      alpha[r] = __expf(mrow[r] - mnew[r]);
      mrow[r] = mnew[r];
    }
#pragma unroll
    for (int r = 0; r < 8; r++) {
      float ps = 0.f;
#pragma unroll
      for (int nt = 0; nt < 4; nt++) {
        float p = __expf(s[nt][r] - mnew[r]);
        s[nt][r] = p; ps += p;
      }
      for (int m = 1; m < 16; m <<= 1) ps += __shfl_xor(ps, m, 32);
      lrow[r] = lrow[r] * alpha[r] + ps;
    }
#pragma unroll
    for (int dn = 0; dn < 4; dn++)
#pragma unroll
      for (int r = 0; r < 8; r++) O[dn][r] *= alpha[r];

    // C-layout P -> LDS -> A-fragment layout
    ushort_t* pb = &Pbuf[wid][0];
#pragma unroll
    for (int nt = 0; nt < 4; nt++)
#pragma unroll
      for (int r = 0; r < 8; r++)
        pb[(r + 8 * hi) * 64 + nt * 16 + lo] = f2bf(s[nt][r]);
    asm volatile("s_wait_dscnt 0" ::: "memory");
    const v4u* pr = (const v4u*)(pb + lo * 64);
    v16bf aP0 = mkfrag(pr[hi], pr[2 + hi]);       // k(t) 0..31 of tile
    v16bf aP1 = mkfrag(pr[4 + hi], pr[6 + hi]);   // k(t) 32..63

#pragma unroll
    for (int dn = 0; dn < 4; dn++) {
      const v4u* vr = (const v4u*)(vbase + (size_t)(dn * 16 + lo) * 1024) + kt * 8;
      v16bf bv0 = mkfrag(vr[hi], vr[2 + hi]);
      v16bf bv1 = mkfrag(vr[4 + hi], vr[6 + hi]);
      O[dn] = wmma_bf16(aP0, bv0, O[dn]);
      O[dn] = wmma_bf16(aP1, bv1, O[dn]);
    }
  }

  // write y (bf16, [B*T, E]) normalized by row sums
#pragma unroll
  for (int dn = 0; dn < 4; dn++)
#pragma unroll
    for (int r = 0; r < 8; r++) {
      int t = q0 + r + 8 * hi;
      int c = hh * 64 + dn * 16 + lo;
      yb[((size_t)b * 1024 + t) * 768 + c] = f2bf(O[dn][r] / lrow[r]);
    }
}

// ---- loss: per-row logsumexp (deterministic 2-stage reduction) -----------

__global__ KLB void loss_kernel(const float* __restrict__ logits,
                                const int* __restrict__ targets,
                                float* __restrict__ lossrow) {
  __shared__ float red[256];
  int row = blockIdx.x;
  const float* lr = logits + (size_t)row * 50304;
  float mx = -1e30f;
  for (int c = threadIdx.x; c < 50304; c += 256) mx = fmaxf(mx, lr[c]);
  red[threadIdx.x] = mx; __syncthreads();
  for (int off = 128; off > 0; off >>= 1) {
    if ((int)threadIdx.x < off)
      red[threadIdx.x] = fmaxf(red[threadIdx.x], red[threadIdx.x + off]);
    __syncthreads();
  }
  mx = red[0]; __syncthreads();
  float s = 0.f;
  for (int c = threadIdx.x; c < 50304; c += 256) s += __expf(lr[c] - mx);
  red[threadIdx.x] = s; __syncthreads();
  for (int off = 128; off > 0; off >>= 1) {
    if ((int)threadIdx.x < off) red[threadIdx.x] += red[threadIdx.x + off];
    __syncthreads();
  }
  if (threadIdx.x == 0) {
    float lse = mx + __logf(red[0]);
    lossrow[row] = lse - lr[targets[row]];
  }
}

__global__ KLB void reduce_loss_kernel(const float* __restrict__ lossrow,
                                       float* __restrict__ out) {
  __shared__ float red[256];
  float s = 0.f;
  for (int i = threadIdx.x; i < 4096; i += 256) s += lossrow[i];
  red[threadIdx.x] = s; __syncthreads();
  for (int off = 128; off > 0; off >>= 1) {
    if ((int)threadIdx.x < off) red[threadIdx.x] += red[threadIdx.x + off];
    __syncthreads();
  }
  if (threadIdx.x == 0) out[0] = red[0] * (1.0f / 4096.0f);
}

// ---- host orchestration ---------------------------------------------------

extern "C" void kernel_launch(void* const* d_in, const int* in_sizes, int n_in,
                              void* d_out, int out_size, void* d_ws, size_t ws_size,
                              hipStream_t stream) {
  (void)in_sizes; (void)n_in; (void)out_size; (void)ws_size;
  const int*   idx      = (const int*)d_in[0];
  const int*   targets  = (const int*)d_in[1];
  const float* wte      = (const float*)d_in[2];
  const float* wpe      = (const float*)d_in[3];
  const float* lnf_s    = (const float*)d_in[4];
  const float* lnf_b    = (const float*)d_in[5];
  const float* ln1_s    = (const float*)d_in[6];
  const float* ln1_b    = (const float*)d_in[7];
  const float* w_attn   = (const float*)d_in[8];
  const float* b_attn   = (const float*)d_in[9];
  const float* w_aproj  = (const float*)d_in[10];
  const float* b_aproj  = (const float*)d_in[11];
  const float* ln2_s    = (const float*)d_in[12];
  const float* ln2_b    = (const float*)d_in[13];
  const float* w_fc     = (const float*)d_in[14];
  const float* b_fc     = (const float*)d_in[15];
  const float* w_mproj  = (const float*)d_in[16];
  const float* b_mproj  = (const float*)d_in[17];

  char* ws = (char*)d_ws;
  size_t off = 0;
  auto alloc = [&](size_t bytes) -> void* {
    void* p = ws + off;
    off = (off + bytes + 255) & ~(size_t)255;
    return p;
  };
  float*    x    = (float*)alloc(4096ull * 768 * 4);
  float*    qkv  = (float*)alloc(4096ull * 2304 * 4);
  ushort_t* h    = (ushort_t*)alloc(4096ull * 768 * 2);
  ushort_t* qb   = (ushort_t*)alloc(48ull * 1024 * 64 * 2);
  ushort_t* kbuf = (ushort_t*)alloc(48ull * 1024 * 64 * 2);
  ushort_t* vtb  = (ushort_t*)alloc(48ull * 64 * 1024 * 2);
  ushort_t* yb   = (ushort_t*)alloc(4096ull * 768 * 2);
  ushort_t* fcb  = (ushort_t*)alloc(4096ull * 3072 * 2);
  ushort_t* xf   = (ushort_t*)alloc(4096ull * 768 * 2);
  ushort_t* watT = (ushort_t*)alloc(2304ull * 768 * 2);
  ushort_t* wapT = (ushort_t*)alloc(768ull * 768 * 2);
  ushort_t* wfcT = (ushort_t*)alloc(3072ull * 768 * 2);
  ushort_t* wmpT = (ushort_t*)alloc(768ull * 3072 * 2);
  ushort_t* wteb = (ushort_t*)alloc(50304ull * 768 * 2);
  float*    lrow = (float*)alloc(4096ull * 4);

  // tied-embedding weights: already [V,E] = Bt layout, just cast
  {
    long n = 50304L * 768;
    cast_kernel<<<(unsigned)((n + 255) / 256), 256, 0, stream>>>(wte, wteb, n);
  }
  embed_kernel<<<(4096 * 768) / 256, 256, 0, stream>>>(idx, wte, wpe, x);

  for (int l = 0; l < 12; l++) {
    castT_kernel<<<(768 * 2304 + 255) / 256, 256, 0, stream>>>(
        w_attn + (size_t)l * 768 * 2304, watT, 768, 2304);
    castT_kernel<<<(768 * 768 + 255) / 256, 256, 0, stream>>>(
        w_aproj + (size_t)l * 768 * 768, wapT, 768, 768);
    castT_kernel<<<(768 * 3072 + 255) / 256, 256, 0, stream>>>(
        w_fc + (size_t)l * 768 * 3072, wfcT, 768, 3072);
    castT_kernel<<<(3072 * 768 + 255) / 256, 256, 0, stream>>>(
        w_mproj + (size_t)l * 3072 * 768, wmpT, 3072, 768);

    ln_kernel<<<4096, 256, 0, stream>>>(x, ln1_s + l * 768, ln1_b + l * 768, h);
    gemm_kernel<<<dim3(2304 / 128, 4096 / 128), 256, 0, stream>>>(
        h, watT, b_attn + (size_t)l * 2304, nullptr, qkv, nullptr,
        4096, 2304, 768, 0);
    qkvsplit_kernel<<<(4096 * 768) / 256, 256, 0, stream>>>(qkv, qb, kbuf, vtb);
    attn_kernel<<<384, 256, 0, stream>>>(qb, kbuf, vtb, yb);
    gemm_kernel<<<dim3(768 / 128, 4096 / 128), 256, 0, stream>>>(
        yb, wapT, b_aproj + (size_t)l * 768, x, x, nullptr,
        4096, 768, 768, 0);
    ln_kernel<<<4096, 256, 0, stream>>>(x, ln2_s + l * 768, ln2_b + l * 768, h);
    gemm_kernel<<<dim3(3072 / 128, 4096 / 128), 256, 0, stream>>>(
        h, wfcT, b_fc + (size_t)l * 3072, nullptr, nullptr, fcb,
        4096, 3072, 768, 1);
    gemm_kernel<<<dim3(768 / 128, 4096 / 128), 256, 0, stream>>>(
        fcb, wmpT, b_mproj + (size_t)l * 768, x, x, nullptr,
        4096, 768, 3072, 0);
  }

  ln_kernel<<<4096, 256, 0, stream>>>(x, lnf_s, lnf_b, xf);

  float* logits = (float*)d_out;
  gemm_kernel<<<dim3(50304 / 128, 4096 / 128), 256, 0, stream>>>(
      xf, wteb, nullptr, nullptr, logits, nullptr, 4096, 50304, 768, 0);

  loss_kernel<<<4096, 256, 0, stream>>>(logits, targets, lrow);
  reduce_loss_kernel<<<1, 256, 0, stream>>>(lrow, logits + (size_t)4096 * 50304);
}
